// MultiHeadAttention_74320114090281
// MI455X (gfx1250) — matrix operand order repaired and checked by reading, not compiled, not run on hardware
//
#include <hip/hip_runtime.h>
#include <hip/hip_bf16.h>
#include <stdint.h>

// ---------------------------------------------------------------------------
// MHA forward for MI455X (gfx1250, wave32, WMMA bf16 16x16x32, f32 accum)
// Round 3: unroll-by-2 on the pipelined GEMM k-loop so the prefetch registers
// ping-pong instead of rotating through v_mov copies. Otherwise unchanged:
// software-pipelined 32x64 GEMM waves, 32-row flash-attention waves with
// hoisted K/V fragments and an LDS C->A layout bounce for P.
// ---------------------------------------------------------------------------

#define NH     16
#define HID    1024
#define DK     64
#define BATCH  4
#define S_LEN  2048

typedef __attribute__((ext_vector_type(16))) __bf16 v16bf;
typedef __attribute__((ext_vector_type(8)))  __bf16 v8bf;
typedef __attribute__((ext_vector_type(4)))  __bf16 v4bf;
typedef __attribute__((ext_vector_type(8)))  float  v8f;

__device__ __forceinline__ v8f wmma_bf16(v16bf a, v16bf b, v8f c) {
    return __builtin_amdgcn_wmma_f32_16x16x32_bf16(false, a, false, b,
                                                   (short)0, c, false, false);
}

// A fragment (16x32 bf16, MxK) from row-major memory.
// lanes 0-15 -> M=lane, K 0..15; lanes 16-31 -> K 16..31 (contiguous)
__device__ __forceinline__ v16bf load_a_rm(const __bf16* base, int stride, int lane) {
    const __bf16* p = base + (size_t)(lane & 15) * stride + ((lane >> 4) << 4);
    v8bf lo = *(const v8bf*)(p);
    v8bf hi = *(const v8bf*)(p + 8);
    return __builtin_shufflevector(lo, hi, 0,1,2,3,4,5,6,7,8,9,10,11,12,13,14,15);
}

// B fragment (32x16 bf16, KxN) from memory stored N-major: R[n][k] = B(k,n).
// lane = n + 16*(k>=16), half = k & 15 -> 16 contiguous elements per lane.
__device__ __forceinline__ v16bf load_b_nm(const __bf16* base, int stride, int lane) {
    const __bf16* p = base + (size_t)(lane & 15) * stride + ((lane >> 4) << 4);
    return *(const v16bf*)p;  // 32B -> two global_load_b128
}

// ---------------------------------------------------------------------------
// Kernel 1: convert X_Q / X_KV  f32 -> bf16
// ---------------------------------------------------------------------------
__global__ void __launch_bounds__(256)
cvt_x_kernel(const float* __restrict__ Xq, const float* __restrict__ Xkv,
             __bf16* __restrict__ Xqb, __bf16* __restrict__ Xkvb) {
    size_t i = ((size_t)blockIdx.x * blockDim.x + threadIdx.x) * 4;
    const float* src = blockIdx.y ? Xkv : Xq;
    __bf16* dst = blockIdx.y ? Xkvb : Xqb;
    float4 v = *(const float4*)(src + i);
    v4bf o;
    o[0] = (__bf16)v.x; o[1] = (__bf16)v.y; o[2] = (__bf16)v.z; o[3] = (__bf16)v.w;
    *(v4bf*)(dst + i) = o;
}

// ---------------------------------------------------------------------------
// Kernel 2: pack weights (1024x1024 f32 row-major [k][n]) into fragment-major
// bf16: Wp[((nt*KC+kc)*32 + lane)*16 + h] = W[kc*32 + 16*(lane>>4) + h][nt*16 + (lane&15)]
// ---------------------------------------------------------------------------
__global__ void __launch_bounds__(256)
pack_w_kernel(const float* __restrict__ Wq, const float* __restrict__ Wk,
              const float* __restrict__ Wv, const float* __restrict__ Wo,
              __bf16* __restrict__ Wp) {
    const int which = blockIdx.z;
    const float* W = (which == 0) ? Wq : (which == 1) ? Wk : (which == 2) ? Wv : Wo;
    __bf16* dst = Wp + (size_t)which * HID * HID;

    int idx = blockIdx.x * blockDim.x + threadIdx.x;
    const int KC = HID / 32;
    int h    = idx & 15;
    int lane = (idx >> 4) & 31;
    int t    = idx >> 9;
    int kc   = t % KC;
    int nt   = t / KC;
    int n = nt * 16 + (lane & 15);
    int k = kc * 32 + ((lane >> 4) << 4) + h;
    dst[idx] = (__bf16)W[(size_t)k * HID + n];
}

// ---------------------------------------------------------------------------
// Software-pipelined 32(M) x 64(N) x 1024(K) tile: 8 accumulators, next-k
// fragments prefetched into registers while current WMMAs execute.
// Unrolled by 2 so the prefetch registers ping-pong (no rotation moves).
// ---------------------------------------------------------------------------
__device__ __forceinline__ void gemm_tile_32x64(
    const __bf16* __restrict__ Abase,   // rows m0..m0+31, col 0, stride HID
    const __bf16* __restrict__ Wn0,     // packed W + nt0*KC*512
    int lane, v8f acc[2][4]) {
    const int KC = HID / 32;
    const size_t fs = (size_t)KC * 512;   // elements between n-tiles

    v16bf a0 = load_a_rm(Abase, HID, lane);
    v16bf a1 = load_a_rm(Abase + 16 * HID, HID, lane);
    const __bf16* wb = Wn0 + (size_t)lane * 16;
    v16bf b0 = *(const v16bf*)(wb);
    v16bf b1 = *(const v16bf*)(wb + fs);
    v16bf b2 = *(const v16bf*)(wb + 2 * fs);
    v16bf b3 = *(const v16bf*)(wb + 3 * fs);

#pragma unroll 2
    for (int kc = 0; kc < KC; ++kc) {
        int kn = (kc + 1 < KC) ? kc + 1 : kc;   // clamped prefetch index
        const __bf16* An = Abase + kn * 32;
        v16bf an0 = load_a_rm(An, HID, lane);
        v16bf an1 = load_a_rm(An + 16 * HID, HID, lane);
        const __bf16* wn = Wn0 + ((size_t)kn * 32 + lane) * 16;
        v16bf bn0 = *(const v16bf*)(wn);
        v16bf bn1 = *(const v16bf*)(wn + fs);
        v16bf bn2 = *(const v16bf*)(wn + 2 * fs);
        v16bf bn3 = *(const v16bf*)(wn + 3 * fs);

        acc[0][0] = wmma_bf16(a0, b0, acc[0][0]);
        acc[0][1] = wmma_bf16(a0, b1, acc[0][1]);
        acc[0][2] = wmma_bf16(a0, b2, acc[0][2]);
        acc[0][3] = wmma_bf16(a0, b3, acc[0][3]);
        acc[1][0] = wmma_bf16(a1, b0, acc[1][0]);
        acc[1][1] = wmma_bf16(a1, b1, acc[1][1]);
        acc[1][2] = wmma_bf16(a1, b2, acc[1][2]);
        acc[1][3] = wmma_bf16(a1, b3, acc[1][3]);

        a0 = an0; a1 = an1; b0 = bn0; b1 = bn1; b2 = bn2; b3 = bn3;
    }
}

// ---------------------------------------------------------------------------
// Kernel 3: fused Q/K/V projection. grid.z selects Q/K/V.
// Wave: 32x64 tile; block: 8 waves = 64x256 tile.
// Q,K -> head-major [b][h][s][64]; V -> transposed [b][h][64][s].
// ---------------------------------------------------------------------------
__global__ void __launch_bounds__(256)
proj_kernel(const __bf16* __restrict__ Xqb, const __bf16* __restrict__ Xkvb,
            const __bf16* __restrict__ Wp, __bf16* __restrict__ QKV) {
    const int lane  = threadIdx.x & 31;
    const int wave  = threadIdx.x >> 5;
    const int which = blockIdx.z;
    const __bf16* X = (which == 0) ? Xqb : Xkvb;
    const __bf16* W = Wp + (size_t)which * HID * HID;
    __bf16* dst = QKV + (size_t)which * ((size_t)BATCH * NH * S_LEN * DK);

    const int KC = HID / 32;
    const int m0 = (blockIdx.x * 2 + (wave >> 2)) * 32;
    const int n0 = (blockIdx.y * 4 + (wave & 3)) * 64;

    v8f acc[2][4] = {};
    gemm_tile_32x64(X + (size_t)m0 * HID, W + (size_t)(n0 >> 4) * KC * 512, lane, acc);

#pragma unroll
    for (int ms = 0; ms < 2; ++ms)
#pragma unroll
        for (int f = 0; f < 4; ++f) {
            int n = n0 + f * 16 + (lane & 15);
            int hh = n >> 6;
            int d  = n & 63;
#pragma unroll
            for (int r = 0; r < 8; ++r) {
                int m  = m0 + ms * 16 + r + ((lane >> 4) << 3);
                int bb = m >> 11;
                int ss = m & (S_LEN - 1);
                size_t di;
                if (which == 2)  // V transposed: [b][h][d][s]
                    di = (((size_t)(bb * NH + hh)) * DK + d) * S_LEN + ss;
                else             // Q/K head-major: [b][h][s][d]
                    di = (((size_t)(bb * NH + hh)) * S_LEN + ss) * DK + d;
                dst[di] = (__bf16)acc[ms][f][r];
            }
        }
}

// ---------------------------------------------------------------------------
// Online softmax update for one 16-row group over a 16x32 score tile.
// ---------------------------------------------------------------------------
__device__ __forceinline__ void online_softmax(
    v8f& sa, v8f& sb, float ma0, float ma1,
    float* mr, float* lr, v8f* og /*[4]*/,
    __bf16 (*prow)[32], int rowbase, int lane) {
    const float sc = 0.125f * 1.44269504088896340736f;  // 1/sqrt(64)*log2(e)
    float rm[8], rs[8];
#pragma unroll
    for (int r = 0; r < 8; ++r) {
        sa[r] = sa[r] * sc + ma0;
        sb[r] = sb[r] * sc + ma1;
        rm[r] = fmaxf(sa[r], sb[r]);
    }
#pragma unroll
    for (int off = 1; off < 16; off <<= 1)
#pragma unroll
        for (int r = 0; r < 8; ++r)
            rm[r] = fmaxf(rm[r], __shfl_xor(rm[r], off, 32));
#pragma unroll
    for (int r = 0; r < 8; ++r) {
        float mn   = fmaxf(mr[r], rm[r]);
        float corr = exp2f(mr[r] - mn);
        mr[r] = mn;
        float p0 = exp2f(sa[r] - mn);
        float p1 = exp2f(sb[r] - mn);
        rs[r] = p0 + p1;
        lr[r] *= corr;
        og[0][r] *= corr; og[1][r] *= corr; og[2][r] *= corr; og[3][r] *= corr;
        int row = rowbase + r + ((lane >> 4) << 3);
        prow[row][lane & 15]        = (__bf16)p0;
        prow[row][16 + (lane & 15)] = (__bf16)p1;
    }
#pragma unroll
    for (int off = 1; off < 16; off <<= 1)
#pragma unroll
        for (int r = 0; r < 8; ++r)
            rs[r] += __shfl_xor(rs[r], off, 32);
#pragma unroll
    for (int r = 0; r < 8; ++r) lr[r] += rs[r];
}

// ---------------------------------------------------------------------------
// Kernel 4: flash attention. One wave per (b, h, 32-row q-tile).
// Sk chunks of 32; 8 S-WMMAs + 8 O-WMMAs per chunk; K and V fragments hoisted
// to the loop head so V latency hides under S-WMMAs + softmax VALU.
// ---------------------------------------------------------------------------
__global__ void __launch_bounds__(128)
attn_kernel(const __bf16* __restrict__ QKV, const uint8_t* __restrict__ mask,
            __bf16* __restrict__ Yp) {
    const int lane = threadIdx.x & 31;
    const int wave = threadIdx.x >> 5;
    const int gw = blockIdx.x * 4 + wave;        // 0 .. B*NH*(S/32)-1
    const int qt = gw & (S_LEN / 32 - 1);
    const int h  = (gw >> 6) & (NH - 1);
    const int b  = gw >> 10;

    const size_t headElems = (size_t)S_LEN * DK;
    const size_t blk = (size_t)BATCH * NH * headElems;
    const __bf16* Qh = QKV + (size_t)(b * NH + h) * headElems + (size_t)qt * 32 * DK;
    const __bf16* Kh = QKV + blk     + (size_t)(b * NH + h) * headElems;
    const __bf16* Vt = QKV + 2 * blk + (size_t)(b * NH + h) * headElems;  // [64][S]
    const uint8_t* mp = mask + (size_t)b * S_LEN;

    __shared__ __attribute__((aligned(32))) __bf16 plds[4][32][32];

    v16bf qa00 = load_a_rm(Qh,                DK, lane);
    v16bf qa01 = load_a_rm(Qh + 32,           DK, lane);
    v16bf qa10 = load_a_rm(Qh + 16 * DK,      DK, lane);
    v16bf qa11 = load_a_rm(Qh + 16 * DK + 32, DK, lane);

    v8f o[2][4] = {};
    float mstat[2][8], lstat[2][8];
#pragma unroll
    for (int g = 0; g < 2; ++g)
#pragma unroll
        for (int r = 0; r < 8; ++r) { mstat[g][r] = -1.0e30f; lstat[g][r] = 0.0f; }

    for (int kb = 0; kb < S_LEN / 32; ++kb) {
        const __bf16* Kt = Kh + (size_t)kb * 32 * DK;
        const __bf16* Vb = Vt + (size_t)kb * 32;

        // hoisted fragment loads (K now, V consumed after softmax)
        v16bf kf0 = load_b_nm(Kt,                DK, lane);
        v16bf kf1 = load_b_nm(Kt + 32,           DK, lane);
        v16bf kf2 = load_b_nm(Kt + 16 * DK,      DK, lane);
        v16bf kf3 = load_b_nm(Kt + 16 * DK + 32, DK, lane);
        v16bf vf0 = load_b_nm(Vb,                S_LEN, lane);
        v16bf vf1 = load_b_nm(Vb + 16 * S_LEN,   S_LEN, lane);
        v16bf vf2 = load_b_nm(Vb + 32 * S_LEN,   S_LEN, lane);
        v16bf vf3 = load_b_nm(Vb + 48 * S_LEN,   S_LEN, lane);
        float ma0 = mp[kb * 32 + (lane & 15)]      ? -1.0e30f : 0.0f;
        float ma1 = mp[kb * 32 + 16 + (lane & 15)] ? -1.0e30f : 0.0f;

        v8f s00 = {}, s01 = {}, s10 = {}, s11 = {};
        s00 = wmma_bf16(qa00, kf0, s00);
        s00 = wmma_bf16(qa01, kf1, s00);
        s01 = wmma_bf16(qa00, kf2, s01);
        s01 = wmma_bf16(qa01, kf3, s01);
        s10 = wmma_bf16(qa10, kf0, s10);
        s10 = wmma_bf16(qa11, kf1, s10);
        s11 = wmma_bf16(qa10, kf2, s11);
        s11 = wmma_bf16(qa11, kf3, s11);

        online_softmax(s00, s01, ma0, ma1, mstat[0], lstat[0], o[0], plds[wave], 0,  lane);
        online_softmax(s10, s11, ma0, ma1, mstat[1], lstat[1], o[1], plds[wave], 16, lane);

        // wave-private LDS round trip (C-layout -> A-layout); same-wave DS is
        // in-order, fence the compiler and wait for the stores.
        asm volatile("s_wait_dscnt 0" ::: "memory");

        const __bf16* pp0 = &plds[wave][lane & 15][(lane >> 4) << 4];
        const __bf16* pp1 = &plds[wave][16 + (lane & 15)][(lane >> 4) << 4];
        v8bf p0lo = *(const v8bf*)(pp0);
        v8bf p0hi = *(const v8bf*)(pp0 + 8);
        v8bf p1lo = *(const v8bf*)(pp1);
        v8bf p1hi = *(const v8bf*)(pp1 + 8);
        v16bf pf0 = __builtin_shufflevector(p0lo, p0hi, 0,1,2,3,4,5,6,7,8,9,10,11,12,13,14,15);
        v16bf pf1 = __builtin_shufflevector(p1lo, p1hi, 0,1,2,3,4,5,6,7,8,9,10,11,12,13,14,15);

        o[0][0] = wmma_bf16(pf0, vf0, o[0][0]);
        o[0][1] = wmma_bf16(pf0, vf1, o[0][1]);
        o[0][2] = wmma_bf16(pf0, vf2, o[0][2]);
        o[0][3] = wmma_bf16(pf0, vf3, o[0][3]);
        o[1][0] = wmma_bf16(pf1, vf0, o[1][0]);
        o[1][1] = wmma_bf16(pf1, vf1, o[1][1]);
        o[1][2] = wmma_bf16(pf1, vf2, o[1][2]);
        o[1][3] = wmma_bf16(pf1, vf3, o[1][3]);

        // keep next iteration's LDS stores behind this iteration's loads
        asm volatile("" ::: "memory");
    }

    // normalize, write Yp[b][s][h*64 + d] (bf16 row-major 8192x1024)
    __bf16* yp = Yp + ((size_t)(b * S_LEN + qt * 32)) * HID + h * DK;
#pragma unroll
    for (int g = 0; g < 2; ++g)
#pragma unroll
        for (int r = 0; r < 8; ++r) {
            int row = g * 16 + r + ((lane >> 4) << 3);
            float inv = 1.0f / lstat[g][r];
            int c = lane & 15;
            yp[(size_t)row * HID + c]      = (__bf16)(o[g][0][r] * inv);
            yp[(size_t)row * HID + 16 + c] = (__bf16)(o[g][1][r] * inv);
            yp[(size_t)row * HID + 32 + c] = (__bf16)(o[g][2][r] * inv);
            yp[(size_t)row * HID + 48 + c] = (__bf16)(o[g][3][r] * inv);
        }
}

// ---------------------------------------------------------------------------
// Kernel 5: output projection  Y = Yp(bf16) @ W_O -> f32 row-major
// ---------------------------------------------------------------------------
__global__ void __launch_bounds__(256)
oproj_kernel(const __bf16* __restrict__ Yp, const __bf16* __restrict__ Wo,
             float* __restrict__ out) {
    const int lane = threadIdx.x & 31;
    const int wave = threadIdx.x >> 5;
    const int KC = HID / 32;
    const int m0 = (blockIdx.x * 2 + (wave >> 2)) * 32;
    const int n0 = (blockIdx.y * 4 + (wave & 3)) * 64;

    v8f acc[2][4] = {};
    gemm_tile_32x64(Yp + (size_t)m0 * HID, Wo + (size_t)(n0 >> 4) * KC * 512, lane, acc);

#pragma unroll
    for (int ms = 0; ms < 2; ++ms)
#pragma unroll
        for (int f = 0; f < 4; ++f) {
            int n = n0 + f * 16 + (lane & 15);
#pragma unroll
            for (int r = 0; r < 8; ++r) {
                int m = m0 + ms * 16 + r + ((lane >> 4) << 3);
                out[(size_t)m * HID + n] = acc[ms][f][r];
            }
        }
}

// ---------------------------------------------------------------------------
// Launch
// ---------------------------------------------------------------------------
extern "C" void kernel_launch(void* const* d_in, const int* in_sizes, int n_in,
                              void* d_out, int out_size, void* d_ws, size_t ws_size,
                              hipStream_t stream) {
    const float*   X_Q  = (const float*)d_in[0];
    const float*   X_KV = (const float*)d_in[1];
    const uint8_t* mask = (const uint8_t*)d_in[2];  // jax bool -> 1 byte/elem
    const float*   W_Q  = (const float*)d_in[3];
    const float*   W_K  = (const float*)d_in[4];
    const float*   W_V  = (const float*)d_in[5];
    const float*   W_O  = (const float*)d_in[6];
    float* out = (float*)d_out;

    // workspace carve-up (total 104 MB)
    char* ws = (char*)d_ws;
    __bf16* Xqb  = (__bf16*)(ws);                              // 16 MB
    __bf16* Xkvb = (__bf16*)(ws + (size_t)16 * 1024 * 1024);   // 16 MB
    __bf16* Wp   = (__bf16*)(ws + (size_t)32 * 1024 * 1024);   //  8 MB (4 weights)
    __bf16* QKV  = (__bf16*)(ws + (size_t)40 * 1024 * 1024);   // 48 MB (Q,K,Vt)
    __bf16* Yp   = (__bf16*)(ws + (size_t)88 * 1024 * 1024);   // 16 MB

    cvt_x_kernel <<<dim3(8192, 2),    256, 0, stream>>>(X_Q, X_KV, Xqb, Xkvb);
    pack_w_kernel<<<dim3(4096, 1, 4), 256, 0, stream>>>(W_Q, W_K, W_V, W_O, Wp);
    proj_kernel  <<<dim3(128, 4, 3),  256, 0, stream>>>(Xqb, Xkvb, Wp, QKV);
    attn_kernel  <<<dim3(1024),       128, 0, stream>>>(QKV, mask, Yp);
    oproj_kernel <<<dim3(128, 4),     256, 0, stream>>>(Yp, Wp + (size_t)3 * HID * HID, out);
}
